// SimpleTransformer_15410342658158
// MI455X (gfx1250) — matrix operand order repaired and checked by reading, not compiled, not run on hardware
//
#include <hip/hip_runtime.h>
#include <hip/hip_bf16.h>

// ---------------------------------------------------------------------------
// MI455X (gfx1250) transformer forward.  All matmuls on WMMA matrix cores,
// bf16 inputs / f32 accumulation (v_wmma_f32_16x16x32_bf16, wave32).
// GEMM tiles are moved global->LDS with GLOBAL_LOAD_ASYNC_TO_LDS_B128
// (ASYNCcnt-tracked, zero VGPR staging -> no spills), double-buffered.
// Weights pre-transposed+converted to bf16 once (L2-resident thereafter).
// ---------------------------------------------------------------------------

#define TB   4
#define TS   1024
#define TD   768
#define TH_  12
#define TDK  64
#define TDFF 3072
#define TV   32000
#define TL   2

#define LDSN 40   // padded LDS row stride (bf16 elems) for 32-wide tiles
#define LQ   72   // padded LDS row stride (bf16 elems) for 64-wide tiles

typedef __attribute__((ext_vector_type(16))) __bf16 bf16x16;
typedef __attribute__((ext_vector_type(8)))  __bf16 bf16x8;
typedef __attribute__((ext_vector_type(8)))  float  f32x8;

__device__ __forceinline__ __bf16 to_bf(float x) { return (__bf16)x; }

// LDS offset of a __shared__ object = low 32 bits of its flat address
// (ISA 10.2: LDS aperture maps addr[31:0] to the wave's LDS).
__device__ __forceinline__ unsigned lds_off(const void* p) {
  return (unsigned)(size_t)p;
}

// Async 16B global->LDS copy, tracked by ASYNCcnt.
__device__ __forceinline__ void async_copy_b128(unsigned dst_lds, const void* src) {
  asm volatile("global_load_async_to_lds_b128 %0, %1, off"
               :: "v"(dst_lds), "v"((unsigned long long)(size_t)src)
               : "memory");
}
__device__ __forceinline__ void wait_async0() {
  asm volatile("s_wait_asynccnt 0" ::: "memory");
}

// A-fragment (16x32, M x K): lane m=lane&15, halves hold K {0-7,16-23}/{8-15,24-31}.
__device__ __forceinline__ bf16x16 load_frag_a(const __bf16* base, int lane, int stride) {
  int m = lane & 15, hf = lane >> 4;
  const __bf16* p = base + m * stride + hf * 8;
  bf16x8 lo = *(const bf16x8*)(p);
  bf16x8 hi = *(const bf16x8*)(p + 16);
  return __builtin_shufflevector(lo, hi, 0,1,2,3,4,5,6,7,8,9,10,11,12,13,14,15);
}

// B-fragment (32x16, K x N): lane n=lane&15 holds column n, K {0-7,16-23}/{8-15,24-31}.
// LDS holds B^T rows: [n][k].
__device__ __forceinline__ bf16x16 load_frag_b(const __bf16* base, int lane, int stride) {
  int n = lane & 15, hf = lane >> 4;
  const __bf16* p = base + n * stride + hf * 8;
  bf16x8 lo = *(const bf16x8*)(p);
  bf16x8 hi = *(const bf16x8*)(p + 16);
  return __builtin_shufflevector(lo, hi, 0,1,2,3,4,5,6,7,8,9,10,11,12,13,14,15);
}

__device__ __forceinline__ f32x8 wmma_bf16(bf16x16 a, bf16x16 b, f32x8 c) {
  return __builtin_amdgcn_wmma_f32_16x16x32_bf16(false, a, false, b, (short)0, c,
                                                 false, false);
}

// ---------------------------------------------------------------------------
// Weight prep: Wt[n][k] = (bf16) W[k][n].  32x32 tiles via LDS.
// ---------------------------------------------------------------------------
__global__ __launch_bounds__(256) void transpose_bf16_kernel(
    const float* __restrict__ W, __bf16* __restrict__ Wt, int K, int N)
{
  __shared__ float tile[32][33];
  const int n0 = blockIdx.x * 32, k0 = blockIdx.y * 32;
  const int tx = threadIdx.x & 31, ty = threadIdx.x >> 5;   // 32 x 8
  #pragma unroll
  for (int i = 0; i < 4; ++i)
    tile[ty + i * 8][tx] = W[(size_t)(k0 + ty + i * 8) * N + n0 + tx];
  __syncthreads();
  #pragma unroll
  for (int i = 0; i < 4; ++i)
    Wt[(size_t)(n0 + ty + i * 8) * K + k0 + tx] = to_bf(tile[tx][ty + i * 8]);
}

// ---------------------------------------------------------------------------
// GEMM: C = act(A[M,K] @ Wt[N,K]^T + bias).  A, Wt in bf16 row-major.
// 128x128 block tile, 8 waves (2x4), wave = 64x32 -> 4x2 accumulators.
// Double-buffered LDS fed by async global->LDS copies (no VGPR staging).
// Output: f32 (Cf) or bf16 (Cb) — exactly one non-null.
// ---------------------------------------------------------------------------
__global__ __launch_bounds__(256) void gemm_bf16_kernel(
    const __bf16* __restrict__ A, const __bf16* __restrict__ Wt,
    const float* __restrict__ bias, float* __restrict__ Cf,
    __bf16* __restrict__ Cb, int M, int N, int K, int relu)
{
  __shared__ __align__(16) __bf16 sA[2][128 * LDSN];
  __shared__ __align__(16) __bf16 sB[2][128 * LDSN];
  const int t = threadIdx.x, lane = t & 31, wave = t >> 5;
  const int m0 = blockIdx.y * 128, n0 = blockIdx.x * 128;
  const int mw = wave >> 2, nw = wave & 3;     // 2 x 4 wave grid
  f32x8 acc[4][2] = {};

  // Per-thread tile coordinates: 1024 16B chunks (A: 512, B: 512) / 256 = 4.
  // Precompute global base pointers (kc advances them) and LDS offsets.
  const __bf16* gsrc[4];
  unsigned      loff[2][4];
  #pragma unroll
  for (int c = 0; c < 4; ++c) {
    const int idx = c * 256 + t;
    const int which = idx >> 9, local = idx & 511;
    const int row = local >> 2, part = local & 3;
    gsrc[c] = (which ? Wt + (size_t)(n0 + row) * K
                     : A  + (size_t)(m0 + row) * K) + part * 8;
    const int e = row * LDSN + part * 8;
    loff[0][c] = lds_off(which ? &sB[0][e] : &sA[0][e]);
    loff[1][c] = lds_off(which ? &sB[1][e] : &sA[1][e]);
  }
  auto issue = [&](int kc, int buf) {
    #pragma unroll
    for (int c = 0; c < 4; ++c)
      async_copy_b128(loff[buf][c], gsrc[c] + kc);
  };

  issue(0, 0);
  wait_async0();
  __syncthreads();

  const int nk = K / 32;
  for (int ki = 0; ki < nk; ++ki) {
    const int cur = ki & 1;
    if (ki + 1 < nk) issue((ki + 1) * 32, cur ^ 1);   // overlaps WMMA below
    bf16x16 bf0 = load_frag_b(sB[cur] + (nw * 32 + 0)  * LDSN, lane, LDSN);
    bf16x16 bf1 = load_frag_b(sB[cur] + (nw * 32 + 16) * LDSN, lane, LDSN);
    #pragma unroll
    for (int i = 0; i < 4; ++i) {
      bf16x16 af = load_frag_a(sA[cur] + (mw * 64 + i * 16) * LDSN, lane, LDSN);
      acc[i][0] = wmma_bf16(af, bf0, acc[i][0]);
      acc[i][1] = wmma_bf16(af, bf1, acc[i][1]);
    }
    if (ki + 1 < nk) wait_async0();
    __syncthreads();
  }

  const int hf = lane >> 4, nl = lane & 15;
  #pragma unroll
  for (int i = 0; i < 4; ++i)
    #pragma unroll
    for (int j = 0; j < 2; ++j) {
      const int col = n0 + nw * 32 + j * 16 + nl;
      const float bb = bias ? bias[col] : 0.0f;
      #pragma unroll
      for (int v = 0; v < 8; ++v) {
        float val = acc[i][j][v] + bb;
        if (relu) val = fmaxf(val, 0.0f);
        const int row = m0 + mw * 64 + i * 16 + hf * 8 + v;
        if (Cf) Cf[(size_t)row * N + col] = val;
        else    Cb[(size_t)row * N + col] = to_bf(val);
      }
    }
}

// ---------------------------------------------------------------------------
// Attention scores: per (b,h), scores[q,k] = inv_temp * Q[q,:] . K[k,:].
// Q,K bf16.  Whole DK=64 staged at once via async copies -> single barrier.
// Only blocks intersecting the causal lower triangle are computed.
// ---------------------------------------------------------------------------
__global__ __launch_bounds__(256) void attn_scores_kernel(
    const __bf16* __restrict__ q, const __bf16* __restrict__ k,
    float* __restrict__ attn)
{
  const int bh = blockIdx.z;
  const int b = bh / TH_, hh = bh % TH_;
  const int m0 = blockIdx.y * 128;     // query tile
  const int n0 = blockIdx.x * 128;     // key tile
  if (n0 > m0 + 127) return;           // fully masked: softmax never reads it

  const __bf16* qptr = q + (size_t)b * TS * TD + hh * TDK;
  const __bf16* kptr = k + (size_t)b * TS * TD + hh * TDK;
  float* aptr = attn + (size_t)bh * TS * TS;

  __shared__ __align__(16) __bf16 sQ[128 * LQ];
  __shared__ __align__(16) __bf16 sK[128 * LQ];
  const int t = threadIdx.x, lane = t & 31, wave = t >> 5;
  const int mw = wave >> 2, nw = wave & 3;
  f32x8 acc[4][2] = {};

  // 2 tiles x 128 rows x 8 chunks(16B) = 2048 chunks / 256 threads = 8
  #pragma unroll
  for (int c = 0; c < 8; ++c) {
    const int idx = c * 256 + t;
    const int which = idx >> 10, local = idx & 1023;
    const int row = local >> 3, part = local & 7;
    const __bf16* src = which ? kptr + (size_t)(n0 + row) * TD + part * 8
                              : qptr + (size_t)(m0 + row) * TD + part * 8;
    async_copy_b128(lds_off((which ? sK : sQ) + row * LQ + part * 8), src);
  }
  wait_async0();
  __syncthreads();

  #pragma unroll
  for (int kc = 0; kc < TDK; kc += 32) {
    bf16x16 bf0 = load_frag_b(sK + (nw * 32 + 0)  * LQ + kc, lane, LQ);
    bf16x16 bf1 = load_frag_b(sK + (nw * 32 + 16) * LQ + kc, lane, LQ);
    #pragma unroll
    for (int i = 0; i < 4; ++i) {
      bf16x16 af = load_frag_a(sQ + (mw * 64 + i * 16) * LQ + kc, lane, LQ);
      acc[i][0] = wmma_bf16(af, bf0, acc[i][0]);
      acc[i][1] = wmma_bf16(af, bf1, acc[i][1]);
    }
  }

  const float inv_temp = 0.125f;   // 1/sqrt(64)
  const int hf = lane >> 4, nl = lane & 15;
  #pragma unroll
  for (int i = 0; i < 4; ++i)
    #pragma unroll
    for (int j = 0; j < 2; ++j) {
      const int col = n0 + nw * 32 + j * 16 + nl;
      #pragma unroll
      for (int v = 0; v < 8; ++v) {
        const int row = m0 + mw * 64 + i * 16 + hf * 8 + v;
        aptr[(size_t)row * TS + col] = acc[i][j][v] * inv_temp;
      }
    }
}

// ---------------------------------------------------------------------------
// In-place causal softmax over one score row; exact zeros above the diagonal
// (matches f32 underflow of exp(-1e9 - max) in the reference).
// ---------------------------------------------------------------------------
__global__ __launch_bounds__(256) void softmax_kernel(float* __restrict__ attn)
{
  const int qi = blockIdx.x % TS;
  float* p = attn + (size_t)blockIdx.x * TS;
  const int len = qi + 1;
  const int t = threadIdx.x;
  __shared__ float red[256];

  float mx = -3.4e38f;
  for (int k = t; k < len; k += 256) mx = fmaxf(mx, p[k]);
  red[t] = mx; __syncthreads();
  for (int s = 128; s > 0; s >>= 1) { if (t < s) red[t] = fmaxf(red[t], red[t + s]); __syncthreads(); }
  mx = red[0]; __syncthreads();

  float ev[4]; int cnt = 0;
  float sum = 0.0f;
  for (int k = t; k < len; k += 256) { float e = expf(p[k] - mx); ev[cnt++] = e; sum += e; }
  red[t] = sum; __syncthreads();
  for (int s = 128; s > 0; s >>= 1) { if (t < s) red[t] += red[t + s]; __syncthreads(); }
  const float inv = 1.0f / red[0];

  cnt = 0;
  for (int k = t; k < len; k += 256) p[k] = ev[cnt++] * inv;
  for (int k = t; k < TS; k += 256) if (k >= len) p[k] = 0.0f;
}

// ---------------------------------------------------------------------------
// O = P @ V per (b,h).  P f32 (converted inline, read once), V bf16.
// 128 q-rows x 64 dims per block; causal k-truncation.  O written in bf16.
// ---------------------------------------------------------------------------
__global__ __launch_bounds__(256) void attn_av_kernel(
    const float* __restrict__ attn, const __bf16* __restrict__ v,
    __bf16* __restrict__ o)
{
  const int bh = blockIdx.y;
  const int b = bh / TH_, hh = bh % TH_;
  const int m0 = blockIdx.x * 128;
  const float*  aptr = attn + (size_t)bh * TS * TS;
  const __bf16* vptr = v + (size_t)b * TS * TD + hh * TDK;
  __bf16*       optr = o + (size_t)b * TS * TD + hh * TDK;

  __shared__ __align__(16) __bf16 sA[128 * LDSN];
  __shared__ __align__(16) __bf16 sV[64 * LDSN];
  const int t = threadIdx.x, lane = t & 31, wave = t >> 5;
  const int mw = wave >> 1, nw = wave & 1;     // 4 x 2 wave grid
  f32x8 acc[2][2] = {};
  const int kend = m0 + 128;                   // keys > m0+127 have P == 0

  for (int kc = 0; kc < kend; kc += 32) {
    // batch all global loads before converting/storing
    float4 ar[4];
    const int col = (t & 7) * 4;
    #pragma unroll
    for (int p = 0; p < 4; ++p)
      ar[p] = *(const float4*)(aptr + (size_t)(m0 + p * 32 + (t >> 3)) * TS + kc + col);
    __bf16 vr[8];
    #pragma unroll
    for (int p = 0; p < 8; ++p) {
      const int idx = p * 256 + t;
      vr[p] = vptr[(size_t)(kc + (idx >> 6)) * TD + (idx & 63)];
    }
    #pragma unroll
    for (int p = 0; p < 4; ++p) {
      __bf16* d = sA + (p * 32 + (t >> 3)) * LDSN + col;
      d[0] = to_bf(ar[p].x); d[1] = to_bf(ar[p].y);
      d[2] = to_bf(ar[p].z); d[3] = to_bf(ar[p].w);
    }
    #pragma unroll
    for (int p = 0; p < 8; ++p) {
      const int idx = p * 256 + t;
      sV[(idx & 63) * LDSN + (idx >> 6)] = vr[p];
    }
    __syncthreads();
    bf16x16 bf0 = load_frag_b(sV + (nw * 32 + 0)  * LDSN, lane, LDSN);
    bf16x16 bf1 = load_frag_b(sV + (nw * 32 + 16) * LDSN, lane, LDSN);
    #pragma unroll
    for (int i = 0; i < 2; ++i) {
      bf16x16 af = load_frag_a(sA + (mw * 32 + i * 16) * LDSN, lane, LDSN);
      acc[i][0] = wmma_bf16(af, bf0, acc[i][0]);
      acc[i][1] = wmma_bf16(af, bf1, acc[i][1]);
    }
    __syncthreads();
  }

  const int hf = lane >> 4, nl = lane & 15;
  #pragma unroll
  for (int i = 0; i < 2; ++i)
    #pragma unroll
    for (int j = 0; j < 2; ++j) {
      const int col = nw * 32 + j * 16 + nl;
      #pragma unroll
      for (int vv = 0; vv < 8; ++vv) {
        const int row = m0 + mw * 32 + i * 16 + hf * 8 + vv;
        optr[(size_t)row * TD + col] = to_bf(acc[i][j][vv]);
      }
    }
}

// ---------------------------------------------------------------------------
// h = LayerNorm(h + t) * g + be ; also refresh bf16 shadow hb.
// ---------------------------------------------------------------------------
__global__ __launch_bounds__(256) void add_ln_kernel(
    float* __restrict__ h, __bf16* __restrict__ hb,
    const float* __restrict__ tb,
    const float* __restrict__ g, const float* __restrict__ be)
{
  const int row = blockIdx.x;
  float* hp = h + (size_t)row * TD;
  __bf16* hbp = hb + (size_t)row * TD;
  const float* tp = tb + (size_t)row * TD;
  const int t = threadIdx.x;
  __shared__ float red[256];

  float y[3]; float s = 0.0f;
  #pragma unroll
  for (int i = 0; i < 3; ++i) { const int c = t + i * 256; y[i] = hp[c] + tp[c]; s += y[i]; }
  red[t] = s; __syncthreads();
  for (int w = 128; w > 0; w >>= 1) { if (t < w) red[t] += red[t + w]; __syncthreads(); }
  const float mean = red[0] * (1.0f / TD); __syncthreads();

  float vs = 0.0f;
  #pragma unroll
  for (int i = 0; i < 3; ++i) { const float d = y[i] - mean; vs += d * d; }
  red[t] = vs; __syncthreads();
  for (int w = 128; w > 0; w >>= 1) { if (t < w) red[t] += red[t + w]; __syncthreads(); }
  const float rstd = rsqrtf(red[0] * (1.0f / TD) + 1e-5f);

  #pragma unroll
  for (int i = 0; i < 3; ++i) {
    const int c = t + i * 256;
    const float out = (y[i] - mean) * rstd * g[c] + be[c];
    hp[c] = out;
    hbp[c] = to_bf(out);
  }
}

// ---------------------------------------------------------------------------
// h = emb[x] * sqrt(D) + pe[s]  (f32 + bf16 shadow)
// ---------------------------------------------------------------------------
__global__ __launch_bounds__(256) void embed_kernel(
    const int* __restrict__ x, const float* __restrict__ emb,
    const float* __restrict__ pe, float* __restrict__ h, __bf16* __restrict__ hb)
{
  const int row = blockIdx.x;        // b*S + s
  const int s = row & (TS - 1);
  const int tok = x[row];
  const float scale = 27.712812921102035f;   // sqrt(768)
  for (int c = threadIdx.x; c < TD; c += 256) {
    const float val = emb[(size_t)tok * TD + c] * scale + pe[(size_t)s * TD + c];
    h[(size_t)row * TD + c] = val;
    hb[(size_t)row * TD + c] = to_bf(val);
  }
}

// ---------------------------------------------------------------------------
extern "C" void kernel_launch(void* const* d_in, const int* in_sizes, int n_in,
                              void* d_out, int out_size, void* d_ws, size_t ws_size,
                              hipStream_t stream)
{
  (void)in_sizes; (void)n_in; (void)out_size; (void)ws_size;

  const int*   x    = (const int*)  d_in[0];
  // d_in[1] = mask (tril) — encoded analytically as causal masking
  const float* emb  = (const float*)d_in[2];
  const float* pe   = (const float*)d_in[3];
  const float* wq   = (const float*)d_in[4];
  const float* bq   = (const float*)d_in[5];
  const float* wk   = (const float*)d_in[6];
  const float* bk   = (const float*)d_in[7];
  const float* wv   = (const float*)d_in[8];
  const float* bv   = (const float*)d_in[9];
  const float* wo   = (const float*)d_in[10];
  const float* bo   = (const float*)d_in[11];
  const float* w1   = (const float*)d_in[12];
  const float* b1   = (const float*)d_in[13];
  const float* w2   = (const float*)d_in[14];
  const float* b2   = (const float*)d_in[15];
  const float* g1   = (const float*)d_in[16];
  const float* be1  = (const float*)d_in[17];
  const float* g2   = (const float*)d_in[18];
  const float* be2  = (const float*)d_in[19];
  const float* wout = (const float*)d_in[20];
  const float* bout = (const float*)d_in[21];

  float* logits = (float*)d_out;                          // B*S*V f32
  float* attn0  = logits + (size_t)TB * TS * TV;          // L*B*H*S*S f32

  const size_t SZH = (size_t)TB * TS * TD;                // 3,145,728
  const size_t DD  = (size_t)TD * TD;
  const size_t DF  = (size_t)TD * TDFF;

  float*  h   = (float*)d_ws;
  float*  tb  = h + SZH;
  __bf16* hb  = (__bf16*)(tb + SZH);
  __bf16* qb  = hb + SZH;
  __bf16* kb  = qb + SZH;
  __bf16* vb  = kb + SZH;
  __bf16* ob  = vb + SZH;
  __bf16* ffb = qb;                    // alias q/k/v/o region (exactly B*S*DFF)
  __bf16* wT  = ob + SZH;              // bf16 transposed weights

  __bf16* wqT[TL]; __bf16* wkT[TL]; __bf16* wvT[TL]; __bf16* woT[TL];
  __bf16* w1T[TL]; __bf16* w2T[TL];
  __bf16* p = wT;
  for (int l = 0; l < TL; ++l) {
    wqT[l] = p; p += DD;  wkT[l] = p; p += DD;
    wvT[l] = p; p += DD;  woT[l] = p; p += DD;
    w1T[l] = p; p += DF;  w2T[l] = p; p += DF;
  }
  __bf16* woutT = p;                                       // V x D

  const int M = TB * TS;                                   // 4096
  const dim3 blk(256);

  // ---- one-time weight transpose/convert (L2-resident bf16 thereafter) ----
  for (int l = 0; l < TL; ++l) {
    transpose_bf16_kernel<<<dim3(TD / 32, TD / 32), blk, 0, stream>>>(
        wq + l * DD, wqT[l], TD, TD);
    transpose_bf16_kernel<<<dim3(TD / 32, TD / 32), blk, 0, stream>>>(
        wk + l * DD, wkT[l], TD, TD);
    transpose_bf16_kernel<<<dim3(TD / 32, TD / 32), blk, 0, stream>>>(
        wv + l * DD, wvT[l], TD, TD);
    transpose_bf16_kernel<<<dim3(TD / 32, TD / 32), blk, 0, stream>>>(
        wo + l * DD, woT[l], TD, TD);
    transpose_bf16_kernel<<<dim3(TDFF / 32, TD / 32), blk, 0, stream>>>(
        w1 + l * DF, w1T[l], TD, TDFF);
    transpose_bf16_kernel<<<dim3(TD / 32, TDFF / 32), blk, 0, stream>>>(
        w2 + l * DF, w2T[l], TDFF, TD);
  }
  transpose_bf16_kernel<<<dim3(TV / 32, TD / 32), blk, 0, stream>>>(
      wout, woutT, TD, TV);

  embed_kernel<<<M, blk, 0, stream>>>(x, emb, pe, h, hb);

  for (int l = 0; l < TL; ++l) {
    float* attnL = attn0 + (size_t)l * TB * TH_ * TS * TS;

    gemm_bf16_kernel<<<dim3(TD / 128, M / 128), blk, 0, stream>>>(
        hb, wqT[l], bq + l * TD, nullptr, qb, M, TD, TD, 0);
    gemm_bf16_kernel<<<dim3(TD / 128, M / 128), blk, 0, stream>>>(
        hb, wkT[l], bk + l * TD, nullptr, kb, M, TD, TD, 0);
    gemm_bf16_kernel<<<dim3(TD / 128, M / 128), blk, 0, stream>>>(
        hb, wvT[l], bv + l * TD, nullptr, vb, M, TD, TD, 0);

    attn_scores_kernel<<<dim3(TS / 128, TS / 128, TB * TH_), blk, 0, stream>>>(
        qb, kb, attnL);
    softmax_kernel<<<TB * TH_ * TS, blk, 0, stream>>>(attnL);
    attn_av_kernel<<<dim3(TS / 128, TB * TH_), blk, 0, stream>>>(attnL, vb, ob);

    gemm_bf16_kernel<<<dim3(TD / 128, M / 128), blk, 0, stream>>>(
        ob, woT[l], bo + l * TD, tb, nullptr, M, TD, TD, 0);
    add_ln_kernel<<<M, blk, 0, stream>>>(h, hb, tb, g1 + l * TD, be1 + l * TD);

    gemm_bf16_kernel<<<dim3(TDFF / 128, M / 128), blk, 0, stream>>>(
        hb, w1T[l], b1 + l * TDFF, nullptr, ffb, M, TDFF, TD, 1);
    gemm_bf16_kernel<<<dim3(TD / 128, M / 128), blk, 0, stream>>>(
        ffb, w2T[l], b2 + l * TD, tb, nullptr, M, TD, TDFF, 0);
    add_ln_kernel<<<M, blk, 0, stream>>>(h, hb, tb, g2 + l * TD, be2 + l * TD);
  }

  gemm_bf16_kernel<<<dim3(TV / 128, M / 128), blk, 0, stream>>>(
      hb, woutT, bout, logits, nullptr, M, TV, TD, 0);
}
